// Agent_70566312673448
// MI455X (gfx1250) — matrix operand order repaired and checked by reading, not compiled, not run on hardware
//
#include <hip/hip_runtime.h>
#include <hip/hip_bf16.h>

// ---------------------------------------------------------------------------
// Problem constants (from reference): N=2048 instances, K=16 senses, H=768.
// ---------------------------------------------------------------------------
#define NI   2048
#define KS   16
#define HD   768
#define H2   1536
#define NT48 48     // 768 / 16 column tiles
#define KT48 48     // 1536 / 32 k tiles
#define KT24 24     // 768 / 32 k tiles

// byte strides inside the packed B image: one k-tile row = 48 nt * 512 elem * 2B
#define B_KT_STRIDE 49152
#define B_NT_STRIDE 1024

typedef __attribute__((ext_vector_type(16))) __bf16 v16bf;
typedef __attribute__((ext_vector_type(8)))  float  v8f;
typedef unsigned short ushort_t;

// fp32 -> bf16 round-to-nearest-even (bit trick, no __bf16 arithmetic needed)
static __device__ __forceinline__ ushort_t f2bf(float f) {
    unsigned u = __float_as_uint(f);
    unsigned r = u + 0x7FFFu + ((u >> 16) & 1u);
    return (ushort_t)(r >> 16);
}

// Position of element A[m][kg] inside the LDS A-fragment image.
// 16-bit A 16x32 WMMA layout: lane = m + 16*h, per-lane 16 contiguous bf16:
//   elems 0..7  = K 8h..8h+7, elems 8..15 = K 16+8h..16+8h+7   (per k-tile of 32)
static __device__ __forceinline__ int frag_pos(int m, int kg) {
    int kt = kg >> 5;
    int kp = kg & 31;
    int h  = (kp >> 3) & 1;
    int e  = (kp & 7) | ((kp >> 4) << 3);
    return (kt * 32 + m + 16 * h) * 16 + e;
}

static __device__ __forceinline__ v8f bwmma(v16bf a, v16bf b, v8f c) {
    // emits v_wmma_f32_16x16x32_bf16
    return __builtin_amdgcn_wmma_f32_16x16x32_bf16(false, a, false, b, (short)0, c,
                                                   false, false);
}

static __device__ __forceinline__ float hrsum(float s) {
    // sum across a 16-lane half (rows live per-half in the C fragment)
    s += __shfl_xor(s, 1);
    s += __shfl_xor(s, 2);
    s += __shfl_xor(s, 4);
    s += __shfl_xor(s, 8);
    return s;
}

// ---------------------------------------------------------------------------
// Kernel 1: pack a row-major fp32 weight [Kdim x 768] into bf16 B-fragments.
// Packed index p = ((kt*NT + nt)*32 + lane)*16 + e ; lane half h holds
// k = kt*32 + 8h + (e&7) + 16*(e>>3), col = nt*16 + (lane&15). Per-lane 32B contiguous.
// ---------------------------------------------------------------------------
__global__ void pack_w_kernel(const float* __restrict__ src,
                              ushort_t* __restrict__ dst, int Kdim) {
    const int Ndim = HD, NTt = HD / 16;
    int total = Kdim * Ndim;
    for (int p = blockIdx.x * blockDim.x + threadIdx.x; p < total;
         p += gridDim.x * blockDim.x) {
        int e    = p & 15;
        int lane = (p >> 4) & 31;
        int nt   = (p >> 9) % NTt;
        int kt   = p / (512 * NTt);
        int h    = lane >> 4;
        int col  = nt * 16 + (lane & 15);
        int k    = kt * 32 + 8 * h + (e & 7) + ((e >> 3) << 4);
        dst[p] = f2bf(src[k * Ndim + col]);
    }
}

// ---------------------------------------------------------------------------
// Pipelined GEMM over KTN k-tiles: acc[6] += A(lds) x B(pack), 6 nt-tiles per
// wave.  Byte-pointer form: one base per stream, affine immediate offsets
// (49152 B per kt, 1024 B per nt) -> minimal per-load address VALU.
// Guard-free main loop with distance-1 prefetch; explicit last-kt epilogue.
// ---------------------------------------------------------------------------
template <int KTN>
static __device__ __forceinline__ void gemm6(v8f acc[6],
                                             const ushort_t* __restrict__ ldsA,
                                             const ushort_t* __restrict__ Bpack,
                                             int lane, int wave) {
    const char* ab = (const char*)ldsA + lane * 32;
    const char* bb = (const char*)Bpack + ((wave * 6) * 32 + lane) * 32;
    v16bf a_cur = *(const v16bf*)ab;
    v16bf b_cur = *(const v16bf*)bb;
#pragma unroll 2
    for (int kt = 0; kt < KTN - 1; ++kt) {
        v16bf a_nxt = *(const v16bf*)(ab + (kt + 1) * 1024);
#pragma unroll
        for (int t = 0; t < 6; ++t) {
            int noff = (t == 5) ? (kt + 1) * B_KT_STRIDE
                                : kt * B_KT_STRIDE + (t + 1) * B_NT_STRIDE;
            v16bf b_nxt = *(const v16bf*)(bb + noff);
            acc[t] = bwmma(a_cur, b_cur, acc[t]);
            b_cur = b_nxt;
        }
        a_cur = a_nxt;
    }
    // epilogue: last k-tile, prefetch only within the row
#pragma unroll
    for (int t = 0; t < 6; ++t) {
        v16bf b_nxt = b_cur;
        if (t < 5)
            b_nxt = *(const v16bf*)(bb + (KTN - 1) * B_KT_STRIDE +
                                    (t + 1) * B_NT_STRIDE);
        acc[t] = bwmma(a_cur, b_cur, acc[t]);
        b_cur = b_nxt;
    }
}

// ---------------------------------------------------------------------------
// Kernel 2: per-instance score.  One WG (256 thr = 8 waves) per 16 instances.
// feat = [pred - mean_k] for decision/gloss -> LDS bf16 A-fragment,
// score = relu(feat @ s1_w + s1_b) . s2_w + s2_b  via WMMA.
// ---------------------------------------------------------------------------
__global__ void scores_kernel(const float* __restrict__ dec,
                              const float* __restrict__ gls,
                              const int*   __restrict__ pred,
                              const ushort_t* __restrict__ packS1,
                              const float* __restrict__ s1_b,
                              const float* __restrict__ s2_w,
                              const float* __restrict__ s2_b,
                              float* __restrict__ scores) {
    __shared__ __align__(32) ushort_t ldsA[16 * H2];
    __shared__ float scoreL[16];
    const int tid = threadIdx.x, lane = tid & 31, wave = tid >> 5;
    const int n0 = blockIdx.x * 16;
    if (tid < 16) scoreL[tid] = 0.f;

    for (int m = 0; m < 16; ++m) {
        int n = n0 + m;
        int pk = pred[n];
        const float* db = dec + (size_t)n * KS * HD;
        const float* gb = gls + (size_t)n * KS * HD;
        for (int c = tid; c < HD; c += 256) {
            float sd = 0.f, sg = 0.f;
#pragma unroll
            for (int k = 0; k < KS; ++k) { sd += db[k * HD + c]; sg += gb[k * HD + c]; }
            float fd = db[pk * HD + c] - sd * 0.0625f;
            float fg = gb[pk * HD + c] - sg * 0.0625f;
            ldsA[frag_pos(m, c)]      = f2bf(fd);
            ldsA[frag_pos(m, c + HD)] = f2bf(fg);
        }
    }
    __syncthreads();

    v8f acc[6];
#pragma unroll
    for (int t = 0; t < 6; ++t)
#pragma unroll
        for (int v = 0; v < 8; ++v) acc[t][v] = 0.f;
    gemm6<KT48>(acc, ldsA, packS1, lane, wave);

    const int h = lane >> 4, ncol = lane & 15;
    float part[8];
#pragma unroll
    for (int v = 0; v < 8; ++v) part[v] = 0.f;
#pragma unroll
    for (int t = 0; t < 6; ++t) {
        int c = (wave * 6 + t) * 16 + ncol;
        float b1 = s1_b[c], w2 = s2_w[c];
#pragma unroll
        for (int v = 0; v < 8; ++v) part[v] += fmaxf(acc[t][v] + b1, 0.f) * w2;
    }
#pragma unroll
    for (int v = 0; v < 8; ++v) {
        float s = hrsum(part[v]);
        if (ncol == 0) atomicAdd(&scoreL[8 * h + v], s);
    }
    __syncthreads();
    if (tid < 16) scores[n0 + tid] = scoreL[tid] + s2_b[0];
}

// ---------------------------------------------------------------------------
// Kernel 3 (1 WG): argmax (lowest-index tie-break) + softmax(probs -> d_out),
// then a1 = sel_mg@W1_w + W1_b and av1 = sel_mg@V1_w + V1_b (tiny GEMVs).
// ---------------------------------------------------------------------------
__global__ void reduce_kernel(const float* __restrict__ scores,
                              const float* __restrict__ dec,
                              const float* __restrict__ gls,
                              const int*   __restrict__ pred,
                              const float* __restrict__ W1_w,
                              const float* __restrict__ W1_b,
                              const float* __restrict__ V1_w,
                              const float* __restrict__ V1_b,
                              float* __restrict__ a1,
                              float* __restrict__ av1,
                              int*   __restrict__ selbuf,
                              float* __restrict__ probs_out) {
    __shared__ float rv[256];
    __shared__ int   ri[256];
    __shared__ float smaxS, ssumS;
    __shared__ int   sselS;
    __shared__ float smg[H2];
    const int tid = threadIdx.x;

    float best = -3.4e38f; int bidx = 0x7fffffff;
    for (int i = tid; i < NI; i += 256) {
        float v = scores[i];
        if (v > best || (v == best && i < bidx)) { best = v; bidx = i; }
    }
    rv[tid] = best; ri[tid] = bidx;
    __syncthreads();
    for (int s = 128; s > 0; s >>= 1) {
        if (tid < s) {
            float v = rv[tid + s]; int j = ri[tid + s];
            if (v > rv[tid] || (v == rv[tid] && j < ri[tid])) { rv[tid] = v; ri[tid] = j; }
        }
        __syncthreads();
    }
    if (tid == 0) { smaxS = rv[0]; sselS = ri[0]; selbuf[0] = ri[0]; }
    __syncthreads();
    const float mx = smaxS;

    float part = 0.f;
    for (int i = tid; i < NI; i += 256) part += __expf(scores[i] - mx);
    rv[tid] = part;
    __syncthreads();
    for (int s = 128; s > 0; s >>= 1) {
        if (tid < s) rv[tid] += rv[tid + s];
        __syncthreads();
    }
    if (tid == 0) ssumS = rv[0];
    __syncthreads();
    const float inv = 1.f / ssumS;
    for (int i = tid; i < NI; i += 256) probs_out[i] = __expf(scores[i] - mx) * inv;

    const int sel = sselS;
    const int pk  = pred[sel];
    const float* db = dec + ((size_t)sel * KS + pk) * HD;
    const float* gb = gls + ((size_t)sel * KS + pk) * HD;
    for (int i = tid; i < H2; i += 256) smg[i] = (i < HD) ? db[i] : gb[i - HD];
    __syncthreads();
    for (int o = tid; o < HD; o += 256) {
        float sa = W1_b[o], sv = V1_b[o];
        for (int c = 0; c < H2; ++c) {
            float x = smg[c];
            sa = fmaf(x, W1_w[c * HD + o], sa);
            sv = fmaf(x, V1_w[c * HD + o], sv);
        }
        a1[o] = sa;
        av1[o] = sv;
    }
}

// ---------------------------------------------------------------------------
// Kernel 4: fused update per instance.  One WG (8 waves) per instance n;
// its 16 sense rows are exactly one 16-row WMMA tile, so the K-softmax is
// WG-local.  Two sequential GEMM passes (P then Q) keep only 48 accumulator
// VGPRs live at a time (no spills); LDS A image is reused as A2 afterwards.
// ---------------------------------------------------------------------------
__global__ void update_kernel(const float* __restrict__ dec,
                              const float* __restrict__ gls,
                              const ushort_t* __restrict__ packW2,
                              const ushort_t* __restrict__ packV2,
                              const ushort_t* __restrict__ packU,
                              const float* __restrict__ W2_b,
                              const float* __restrict__ Wa_w,
                              const float* __restrict__ V2_b,
                              const float* __restrict__ U_b,
                              const float* __restrict__ cls_w,
                              const float* __restrict__ cls_b,
                              const float* __restrict__ a1,
                              const float* __restrict__ av1,
                              const int*   __restrict__ selbuf,
                              float* __restrict__ out) {
    __shared__ __align__(32) ushort_t ldsA[16 * H2];  // later reused as A2 (16x768)
    __shared__ float logitL[16];
    __shared__ float attnL[16];
    __shared__ float outL[32];
    const int tid = threadIdx.x, lane = tid & 31, wave = tid >> 5;
    const int n = blockIdx.x;
    if (tid < 16) logitL[tid] = 0.f;
    if (tid < 32) outL[tid] = 0.f;

    const float* db = dec + (size_t)n * KS * HD;
    const float* gb = gls + (size_t)n * KS * HD;

    // Stage 1: tgt_mg tile [16 x 1536] -> bf16 A fragment in LDS
    for (int idx = tid; idx < 16 * H2; idx += 256) {
        int m = idx / H2;
        int c = idx - m * H2;
        float v = (c < HD) ? db[m * HD + c] : gb[m * HD + (c - HD)];
        ldsA[frag_pos(m, c)] = f2bf(v);
    }
    __syncthreads();

    const int h = lane >> 4, ncol = lane & 15;

    // Pass P: P = tgt @ W2, consumed immediately by the tanh . Wa reduction
    {
        v8f accP[6];
#pragma unroll
        for (int t = 0; t < 6; ++t)
#pragma unroll
            for (int v = 0; v < 8; ++v) accP[t][v] = 0.f;
        gemm6<KT48>(accP, ldsA, packW2, lane, wave);

        float part[8];
#pragma unroll
        for (int v = 0; v < 8; ++v) part[v] = 0.f;
#pragma unroll
        for (int t = 0; t < 6; ++t) {
            int c = (wave * 6 + t) * 16 + ncol;
            float a1c = a1[c] + W2_b[c];
            float wac = Wa_w[c];
#pragma unroll
            for (int v = 0; v < 8; ++v) part[v] += tanhf(a1c + accP[t][v]) * wac;
        }
#pragma unroll
        for (int v = 0; v < 8; ++v) {
            float s = hrsum(part[v]);
            if (ncol == 0) atomicAdd(&logitL[8 * h + v], s);
        }
    }

    // Pass Q: Q = tgt @ V2 (reuses the accumulator registers of pass P)
    v8f accQ[6];
#pragma unroll
    for (int t = 0; t < 6; ++t)
#pragma unroll
        for (int v = 0; v < 8; ++v) accQ[t][v] = 0.f;
    gemm6<KT48>(accQ, ldsA, packV2, lane, wave);

    __syncthreads();  // all ldsA reads + logitL adds complete

    // Stage 3b: dv = relu(av1 + Q) -> A2 fragment (overwrites ldsA front);
    // tid 0 folds the K=16 attention softmax in the same phase
    // (Wa_b is a constant shift -> softmax-invariant).
#pragma unroll
    for (int t = 0; t < 6; ++t) {
        int c = (wave * 6 + t) * 16 + ncol;
        float av1c = av1[c] + V2_b[c];
#pragma unroll
        for (int v = 0; v < 8; ++v) {
            int m = 8 * h + v;
            float dv = fmaxf(av1c + accQ[t][v], 0.f);
            ldsA[frag_pos(m, c)] = f2bf(dv);  // A2 image: k dim = 768
        }
    }
    if (tid == 0) {
        float mx = -3.4e38f;
        for (int m = 0; m < 16; ++m) mx = fmaxf(mx, logitL[m]);
        float sm = 0.f;
        float e[16];
        for (int m = 0; m < 16; ++m) { e[m] = __expf(logitL[m] - mx); sm += e[m]; }
        float inv = 1.f / sm;
        for (int m = 0; m < 16; ++m) attnL[m] = e[m] * inv;
    }
    __syncthreads();

    // Stage 5: delta = A2 @ U   ([16x768] x [768x768])
    v8f accD[6];
#pragma unroll
    for (int t = 0; t < 6; ++t)
#pragma unroll
        for (int v = 0; v < 8; ++v) accD[t][v] = 0.f;
    gemm6<KT24>(accD, ldsA, packU, lane, wave);

    // Stage 6: updated = dec + attn*(delta + U_b)  (sel row kept) ; cls logits
    const int sel = selbuf[0];
    float p0[8], p1[8];
#pragma unroll
    for (int v = 0; v < 8; ++v) { p0[v] = 0.f; p1[v] = 0.f; }
#pragma unroll
    for (int t = 0; t < 6; ++t) {
        int c = (wave * 6 + t) * 16 + ncol;
        float ub = U_b[c], c0 = cls_w[c * 2 + 0], c1 = cls_w[c * 2 + 1];
#pragma unroll
        for (int v = 0; v < 8; ++v) {
            int m = 8 * h + v;
            float scale = (n == sel) ? 0.f : attnL[m];
            float upd = db[m * HD + c] + scale * (accD[t][v] + ub);
            p0[v] = fmaf(upd, c0, p0[v]);
            p1[v] = fmaf(upd, c1, p1[v]);
        }
    }
#pragma unroll
    for (int v = 0; v < 8; ++v) {
        float s0 = hrsum(p0[v]);
        float s1 = hrsum(p1[v]);
        if (ncol == 0) {
            int m = 8 * h + v;
            atomicAdd(&outL[2 * m + 0], s0);
            atomicAdd(&outL[2 * m + 1], s1);
        }
    }
    __syncthreads();
    if (tid < 32) {
        int m = tid >> 1, j = tid & 1;
        out[((size_t)n * KS + m) * 2 + j] = outL[tid] + cls_b[j];
    }
}

// ---------------------------------------------------------------------------
extern "C" void kernel_launch(void* const* d_in, const int* in_sizes, int n_in,
                              void* d_out, int out_size, void* d_ws, size_t ws_size,
                              hipStream_t stream) {
    (void)in_sizes; (void)n_in; (void)out_size; (void)ws_size;
    const float* dec   = (const float*)d_in[0];
    const float* gls   = (const float*)d_in[1];
    const int*   pred  = (const int*)  d_in[2];
    const float* s1_w  = (const float*)d_in[3];
    const float* s1_b  = (const float*)d_in[4];
    const float* s2_w  = (const float*)d_in[5];
    const float* s2_b  = (const float*)d_in[6];
    const float* W1_w  = (const float*)d_in[7];
    const float* W1_b  = (const float*)d_in[8];
    const float* W2_w  = (const float*)d_in[9];
    const float* W2_b  = (const float*)d_in[10];
    const float* Wa_w  = (const float*)d_in[11];
    // d_in[12] = Wa_b : constant shift of attn logits, softmax-invariant
    const float* V1_w  = (const float*)d_in[13];
    const float* V1_b  = (const float*)d_in[14];
    const float* V2_w  = (const float*)d_in[15];
    const float* V2_b  = (const float*)d_in[16];
    const float* U_w   = (const float*)d_in[17];
    const float* U_b   = (const float*)d_in[18];
    const float* cls_w = (const float*)d_in[19];
    const float* cls_b = (const float*)d_in[20];

    float* out = (float*)d_out;
    float* probs_out = out + (size_t)NI * KS * 2;  // 65536

    // workspace layout
    ushort_t* packS1 = (ushort_t*)d_ws;
    ushort_t* packW2 = packS1 + (size_t)H2 * HD;
    ushort_t* packV2 = packW2 + (size_t)H2 * HD;
    ushort_t* packU  = packV2 + (size_t)H2 * HD;
    float*    scores = (float*)(packU + (size_t)HD * HD);
    int*      selbuf = (int*)(scores + NI);
    float*    a1     = (float*)(selbuf + 8);
    float*    av1    = a1 + HD;

    hipLaunchKernelGGL(pack_w_kernel, dim3(2048), dim3(256), 0, stream, s1_w, packS1, H2);
    hipLaunchKernelGGL(pack_w_kernel, dim3(2048), dim3(256), 0, stream, W2_w, packW2, H2);
    hipLaunchKernelGGL(pack_w_kernel, dim3(2048), dim3(256), 0, stream, V2_w, packV2, H2);
    hipLaunchKernelGGL(pack_w_kernel, dim3(1024), dim3(256), 0, stream, U_w,  packU,  HD);

    hipLaunchKernelGGL(scores_kernel, dim3(NI / 16), dim3(256), 0, stream,
                       dec, gls, pred, packS1, s1_b, s2_w, s2_b, scores);

    hipLaunchKernelGGL(reduce_kernel, dim3(1), dim3(256), 0, stream,
                       scores, dec, gls, pred, W1_w, W1_b, V1_w, V1_b,
                       a1, av1, selbuf, probs_out);

    hipLaunchKernelGGL(update_kernel, dim3(NI), dim3(256), 0, stream,
                       dec, gls, packW2, packV2, packU, W2_b, Wa_w, V2_b,
                       U_b, cls_w, cls_b, a1, av1, selbuf, out);
}
